// DynamicTransformerBlock_87187836109330
// MI455X (gfx1250) — compile-verified, matrix operand order checked by reading, not run on hardware
//
#include <hip/hip_runtime.h>
#include <math.h>

// ---------------- problem constants ----------------
#define Bsz   8
#define Tseq  1024
#define Edim  1024
#define Hh    16
#define HDd   64
#define MLPd  4096
#define Mrows (Bsz * Tseq)   // 8192

typedef __attribute__((ext_vector_type(16))) __bf16 v16bf;
typedef __attribute__((ext_vector_type(8)))  float  v8f;

#define WMMA_BF16(a, b, c) \
  __builtin_amdgcn_wmma_f32_16x16x32_bf16(false, (a), false, (b), (short)0, (c), false, false)

union FragU { v16bf v; ulonglong2 q[2]; };

// Load a 16x32 bf16 operand fragment (A or B) per the CDNA5 16-bit WMMA layout:
// lane L holds row/col (L & 15); K chunks at offset ((L>>4)*8) and +16.
__device__ __forceinline__ v16bf load_frag(const __bf16* __restrict__ base,
                                           int ld, int rc0, int k0, int lane) {
  const __bf16* p = base + (size_t)(rc0 + (lane & 15)) * (size_t)ld
                         + (size_t)(k0 + ((lane >> 4) << 3));
  FragU f;
  f.q[0] = *reinterpret_cast<const ulonglong2*>(p);
  f.q[1] = *reinterpret_cast<const ulonglong2*>(p + 16);
  return f.v;
}

// Fragment load from a per-lane base pointer (already lane-offset).
__device__ __forceinline__ v16bf load_frag_p(const __bf16* p) {
  FragU f;
  f.q[0] = *reinterpret_cast<const ulonglong2*>(p);
  f.q[1] = *reinterpret_cast<const ulonglong2*>(p + 16);
  return f.v;
}

__device__ __forceinline__ v16bf load_frag_lds(const __bf16* p_lds, int lane) {
  const __bf16* p = p_lds + (lane & 15) * 32 + ((lane >> 4) << 3);
  FragU f;
  f.q[0] = *reinterpret_cast<const ulonglong2*>(p);
  f.q[1] = *reinterpret_cast<const ulonglong2*>(p + 16);
  return f.v;
}

// ---------------- fp32 -> bf16 cast (x4 vectorized) ----------------
__global__ __launch_bounds__(256) void cast4_kernel(const float4* __restrict__ in,
                                                    ushort4* __restrict__ out, int n4) {
  int i = blockIdx.x * 256 + threadIdx.x;
  if (i >= n4) return;
  float4 v = in[i];
  union { ushort4 u; __bf16 h[4]; } t;
  t.h[0] = (__bf16)v.x; t.h[1] = (__bf16)v.y; t.h[2] = (__bf16)v.z; t.h[3] = (__bf16)v.w;
  out[i] = t.u;
}

// ---------------- LayerNorm (row of 1024) fused with bf16 cast ----------------
__global__ __launch_bounds__(256) void ln_kernel(const float* __restrict__ x,
                                                 const float* __restrict__ w,
                                                 const float* __restrict__ b,
                                                 __bf16* __restrict__ out) {
  __shared__ float s_red[16];
  int row = blockIdx.x;
  int t = threadIdx.x;
  const float4* xr = reinterpret_cast<const float4*>(x + (size_t)row * Edim);
  float4 v = xr[t];
  float s  = v.x + v.y + v.z + v.w;
  float sq = v.x * v.x + v.y * v.y + v.z * v.z + v.w * v.w;
#pragma unroll
  for (int off = 16; off; off >>= 1) { s += __shfl_down(s, off); sq += __shfl_down(sq, off); }
  if ((t & 31) == 0) { s_red[t >> 5] = s; s_red[8 + (t >> 5)] = sq; }
  __syncthreads();
  float ts = 0.f, tq = 0.f;
#pragma unroll
  for (int i = 0; i < 8; ++i) { ts += s_red[i]; tq += s_red[8 + i]; }
  float mean = ts * (1.f / Edim);
  float var  = tq * (1.f / Edim) - mean * mean;
  float rstd = rsqrtf(var + 1e-5f);
  const float4* wr = reinterpret_cast<const float4*>(w);
  const float4* br = reinterpret_cast<const float4*>(b);
  float4 wv = wr[t], bv = br[t];
  __bf16* o = out + (size_t)row * Edim + t * 4;
  o[0] = (__bf16)((v.x - mean) * rstd * wv.x + bv.x);
  o[1] = (__bf16)((v.y - mean) * rstd * wv.y + bv.y);
  o[2] = (__bf16)((v.z - mean) * rstd * wv.z + bv.z);
  o[3] = (__bf16)((v.w - mean) * rstd * wv.w + bv.w);
}

// ---------------- generic bf16 WMMA GEMM:  C = A[M,K] * W[N,K]^T + bias ----------------
// 64x32 tile per wave (4 M-frags x 2 N-frags), manually ping-ponged double buffer:
// no register copies; each half-step issues 6 b128-pair loads under 8 WMMAs.
// mode 0: out bf16        mode 1: out fp32 = resid + C        mode 2: out bf16 = gelu(C)

#define LOAD_SET(AF, BF, KOFF)                                            \
  {                                                                       \
    _Pragma("unroll") for (int ni = 0; ni < 2; ++ni)                      \
        BF[ni] = load_frag_p(pb[ni] + (KOFF));                            \
    _Pragma("unroll") for (int mi = 0; mi < 4; ++mi)                      \
        AF[mi] = load_frag_p(pa[mi] + (KOFF));                            \
  }

#define WMMA_SET(AF, BF)                                                  \
  {                                                                       \
    _Pragma("unroll") for (int mi = 0; mi < 4; ++mi)                      \
        _Pragma("unroll") for (int ni = 0; ni < 2; ++ni)                  \
            acc[mi][ni] = WMMA_BF16(AF[mi], BF[ni], acc[mi][ni]);         \
  }

__global__ __launch_bounds__(256) void gemm_bf16_kernel(
    const __bf16* __restrict__ A, const __bf16* __restrict__ W,
    const float* __restrict__ bias, const float* __restrict__ resid,
    float* __restrict__ outF, __bf16* __restrict__ outB,
    int M, int N, int K, int mode) {
  int lane = threadIdx.x & 31;
  int wave = threadIdx.x >> 5;
  int tile = blockIdx.x * 8 + wave;
  int ntn  = N >> 5;                 // N tiles of 32
  int tm = tile / ntn, tn = tile % ntn;
  int m0 = tm << 6, n0 = tn << 5;    // 64 rows x 32 cols per wave
  if (m0 >= M) return;

  // per-lane fragment base pointers
  int lrow  = lane & 15;
  int lkoff = (lane >> 4) << 3;
  const __bf16* pa[4];
  const __bf16* pb[2];
#pragma unroll
  for (int mi = 0; mi < 4; ++mi) pa[mi] = A + (size_t)(m0 + mi * 16 + lrow) * K + lkoff;
#pragma unroll
  for (int ni = 0; ni < 2; ++ni) pb[ni] = W + (size_t)(n0 + ni * 16 + lrow) * K + lkoff;

  v8f acc[4][2];
#pragma unroll
  for (int mi = 0; mi < 4; ++mi)
#pragma unroll
    for (int ni = 0; ni < 2; ++ni) acc[mi][ni] = (v8f){};

  // software-pipelined K loop, unrolled x2 with ping-pong buffers (K/32 is even)
  v16bf af0[4], bf0[2], af1[4], bf1[2];
  LOAD_SET(af0, bf0, 0)
  int k0 = 32;
  for (; k0 < K - 32; k0 += 64) {
    LOAD_SET(af1, bf1, k0)
    WMMA_SET(af0, bf0)
    LOAD_SET(af0, bf0, k0 + 32)
    WMMA_SET(af1, bf1)
  }
  LOAD_SET(af1, bf1, k0)
  WMMA_SET(af0, bf0)
  WMMA_SET(af1, bf1)

  int colA = n0 + lrow;
  int colB = colA + 16;
  float biasA = bias[colA], biasB = bias[colB];
  int rbase = m0 + lkoff;
  const float kGelu = 0.70710678118654752f;
#pragma unroll
  for (int mi = 0; mi < 4; ++mi) {
#pragma unroll
    for (int i = 0; i < 8; ++i) {
      int row = rbase + mi * 16 + i;
      float v0 = acc[mi][0][i] + biasA;
      float v1 = acc[mi][1][i] + biasB;
      size_t i0 = (size_t)row * N + colA;
      size_t i1 = (size_t)row * N + colB;
      if (mode == 0) {
        outB[i0] = (__bf16)v0; outB[i1] = (__bf16)v1;
      } else if (mode == 1) {
        outF[i0] = resid[i0] + v0; outF[i1] = resid[i1] + v1;
      } else {
        outB[i0] = (__bf16)(0.5f * v0 * (1.f + erff(v0 * kGelu)));
        outB[i1] = (__bf16)(0.5f * v1 * (1.f + erff(v1 * kGelu)));
      }
    }
  }
}

// ---------------- V transpose: qkv[B,T,3E] -> vt[B,H,HD,T] ----------------
__global__ __launch_bounds__(256) void vtrans_kernel(const __bf16* __restrict__ qkv,
                                                     __bf16* __restrict__ vt) {
  size_t idx = (size_t)blockIdx.x * 256 + threadIdx.x;
  int t  = (int)(idx & (Tseq - 1));
  int d  = (int)((idx >> 10) & (HDd - 1));
  int bh = (int)(idx >> 16);
  int b = bh >> 4, h = bh & 15;
  vt[idx] = qkv[((size_t)(b * Tseq + t)) * (3 * Edim) + 2 * Edim + h * HDd + d];
}

// ---------------- flash attention: 1 wave per (b,h,16-query tile) ----------------
__global__ __launch_bounds__(32) void attn_kernel(const __bf16* __restrict__ qkv,
                                                  const __bf16* __restrict__ vt,
                                                  __bf16* __restrict__ o) {
  __shared__ __bf16 p_lds[16 * 32];
  int lane = threadIdx.x;
  int qt = blockIdx.x & 63;
  int bh = blockIdx.x >> 6;
  int b = bh >> 4, h = bh & 15;
  int q0 = qt << 4;

  const __bf16* qbase = qkv + ((size_t)(b * Tseq + q0)) * (3 * Edim) + h * HDd;
  const __bf16* kbase = qkv + ((size_t)b * Tseq) * (3 * Edim) + Edim + h * HDd;
  const __bf16* vbase = vt + (size_t)bh * HDd * Tseq;

  v16bf aq0 = load_frag(qbase, 3 * Edim, 0, 0,  lane);
  v16bf aq1 = load_frag(qbase, 3 * Edim, 0, 32, lane);

  v8f o0 = {}, o1 = {}, o2 = {}, o3 = {};
  v8f m, l;
#pragma unroll
  for (int i = 0; i < 8; ++i) { m[i] = -3.0e38f; l[i] = 0.f; }

  for (int kt = 0; kt < Tseq; kt += 32) {
    // S = Q K^T * scale  (two 16x16 tiles covering 32 keys, K-dim = 64)
    v16bf bk00 = load_frag(kbase, 3 * Edim, kt,      0,  lane);
    v16bf bk01 = load_frag(kbase, 3 * Edim, kt,      32, lane);
    v16bf bk10 = load_frag(kbase, 3 * Edim, kt + 16, 0,  lane);
    v16bf bk11 = load_frag(kbase, 3 * Edim, kt + 16, 32, lane);
    v8f s0 = {}, s1 = {};
    s0 = WMMA_BF16(aq0, bk00, s0);
    s0 = WMMA_BF16(aq1, bk01, s0);
    s1 = WMMA_BF16(aq0, bk10, s1);
    s1 = WMMA_BF16(aq1, bk11, s1);

    // online softmax (row stats via butterfly over the 16 lanes of each half)
    v8f mn = m, scale;
#pragma unroll
    for (int i = 0; i < 8; ++i) {
      s0[i] *= 0.125f; s1[i] *= 0.125f;
      float t = fmaxf(s0[i], s1[i]);
      t = fmaxf(t, __shfl_xor(t, 1));
      t = fmaxf(t, __shfl_xor(t, 2));
      t = fmaxf(t, __shfl_xor(t, 4));
      t = fmaxf(t, __shfl_xor(t, 8));
      mn[i] = fmaxf(mn[i], t);
    }
#pragma unroll
    for (int i = 0; i < 8; ++i) {
      scale[i] = __expf(m[i] - mn[i]);
      s0[i] = __expf(s0[i] - mn[i]);
      s1[i] = __expf(s1[i] - mn[i]);
      float r = s0[i] + s1[i];
      r += __shfl_xor(r, 1);
      r += __shfl_xor(r, 2);
      r += __shfl_xor(r, 4);
      r += __shfl_xor(r, 8);
      l[i] = l[i] * scale[i] + r;
      m[i] = mn[i];
    }
#pragma unroll
    for (int i = 0; i < 8; ++i) {
      o0[i] *= scale[i]; o1[i] *= scale[i]; o2[i] *= scale[i]; o3[i] *= scale[i];
    }

    // P (C layout) -> LDS row-major 16x32 -> A fragment
    int prow = (lane >> 4) << 3;
    int pcol = lane & 15;
    __syncthreads();
#pragma unroll
    for (int i = 0; i < 8; ++i) {
      p_lds[(prow + i) * 32 + pcol]      = (__bf16)s0[i];
      p_lds[(prow + i) * 32 + pcol + 16] = (__bf16)s1[i];
    }
    __syncthreads();
    v16bf pf = load_frag_lds(p_lds, lane);

    // O += P * V   (B operand: col = d, K = tokens, from V^T)
    v16bf bv0 = load_frag(vbase, Tseq, 0,  kt, lane);
    v16bf bv1 = load_frag(vbase, Tseq, 16, kt, lane);
    v16bf bv2 = load_frag(vbase, Tseq, 32, kt, lane);
    v16bf bv3 = load_frag(vbase, Tseq, 48, kt, lane);
    o0 = WMMA_BF16(pf, bv0, o0);
    o1 = WMMA_BF16(pf, bv1, o1);
    o2 = WMMA_BF16(pf, bv2, o2);
    o3 = WMMA_BF16(pf, bv3, o3);
  }

  v8f inv;
#pragma unroll
  for (int i = 0; i < 8; ++i) inv[i] = 1.f / l[i];
  int d0 = lane & 15;
  int rbase = q0 + ((lane >> 4) << 3);
#pragma unroll
  for (int i = 0; i < 8; ++i) {
    int row = rbase + i;
    size_t off = ((size_t)(b * Tseq + row)) * Edim + h * HDd;
    o[off + d0]      = (__bf16)(o0[i] * inv[i]);
    o[off + 16 + d0] = (__bf16)(o1[i] * inv[i]);
    o[off + 32 + d0] = (__bf16)(o2[i] * inv[i]);
    o[off + 48 + d0] = (__bf16)(o3[i] * inv[i]);
  }
}

// ---------------- host side ----------------
extern "C" void kernel_launch(void* const* d_in, const int* in_sizes, int n_in,
                              void* d_out, int out_size, void* d_ws, size_t ws_size,
                              hipStream_t stream) {
  const float* x      = (const float*)d_in[0];
  const float* ln1_w  = (const float*)d_in[1];
  const float* ln1_b  = (const float*)d_in[2];
  const float* qkv_w  = (const float*)d_in[3];
  const float* qkv_b  = (const float*)d_in[4];
  const float* proj_w = (const float*)d_in[5];
  const float* proj_b = (const float*)d_in[6];
  const float* ln2_w  = (const float*)d_in[7];
  const float* ln2_b  = (const float*)d_in[8];
  const float* fc1_w  = (const float*)d_in[9];
  const float* fc1_b  = (const float*)d_in[10];
  const float* fc2_w  = (const float*)d_in[11];
  const float* fc2_b  = (const float*)d_in[12];
  float* out = (float*)d_out;

  // workspace carve-up (256B aligned)
  size_t off = 0;
  auto alloc = [&](size_t bytes) {
    size_t o = off;
    off = (off + bytes + 255) & ~(size_t)255;
    return (char*)d_ws + o;
  };
  __bf16* wq  = (__bf16*)alloc((size_t)3 * Edim * Edim * 2);
  __bf16* wp  = (__bf16*)alloc((size_t)Edim * Edim * 2);
  __bf16* w1  = (__bf16*)alloc((size_t)MLPd * Edim * 2);
  __bf16* w2  = (__bf16*)alloc((size_t)Edim * MLPd * 2);
  __bf16* hbf = (__bf16*)alloc((size_t)Mrows * Edim * 2);
  __bf16* qkvb= (__bf16*)alloc((size_t)Mrows * 3 * Edim * 2);
  __bf16* vtb = (__bf16*)alloc((size_t)Bsz * Hh * HDd * Tseq * 2);
  __bf16* ob  = (__bf16*)alloc((size_t)Mrows * Edim * 2);
  float*  x1  = (float*) alloc((size_t)Mrows * Edim * 4);
  __bf16* act = (__bf16*)alloc((size_t)Mrows * MLPd * 2);

  // 1) cast weights to bf16 (every call; no caching)
  {
    int n;
    n = 3 * Edim * Edim; cast4_kernel<<<n / 1024, 256, 0, stream>>>((const float4*)qkv_w,  (ushort4*)wq, n / 4);
    n = Edim * Edim;     cast4_kernel<<<n / 1024, 256, 0, stream>>>((const float4*)proj_w, (ushort4*)wp, n / 4);
    n = MLPd * Edim;     cast4_kernel<<<n / 1024, 256, 0, stream>>>((const float4*)fc1_w,  (ushort4*)w1, n / 4);
    n = Edim * MLPd;     cast4_kernel<<<n / 1024, 256, 0, stream>>>((const float4*)fc2_w,  (ushort4*)w2, n / 4);
  }

  // 2) LN1 -> bf16
  ln_kernel<<<Mrows, 256, 0, stream>>>(x, ln1_w, ln1_b, hbf);

  // 3) QKV GEMM: [8192,1024] x [3072,1024]^T -> bf16   (64x32 per wave)
  gemm_bf16_kernel<<<(Mrows / 64) * (3 * Edim / 32) / 8, 256, 0, stream>>>(
      hbf, wq, qkv_b, nullptr, nullptr, qkvb, Mrows, 3 * Edim, Edim, 0);

  // 4) V transpose -> [B,H,HD,T]
  vtrans_kernel<<<(Bsz * Hh * HDd * Tseq) / 256, 256, 0, stream>>>(qkvb, vtb);

  // 5) attention -> o bf16 [B,T,E]
  attn_kernel<<<Bsz * Hh * (Tseq / 16), 32, 0, stream>>>(qkvb, vtb, ob);

  // 6) proj GEMM + residual -> x1 fp32
  gemm_bf16_kernel<<<(Mrows / 64) * (Edim / 32) / 8, 256, 0, stream>>>(
      ob, wp, proj_b, x, x1, nullptr, Mrows, Edim, Edim, 1);

  // 7) LN2 -> bf16
  ln_kernel<<<Mrows, 256, 0, stream>>>(x1, ln2_w, ln2_b, hbf);

  // 8) fc1 GEMM + exact GELU -> bf16
  gemm_bf16_kernel<<<(Mrows / 64) * (MLPd / 32) / 8, 256, 0, stream>>>(
      hbf, w1, fc1_b, nullptr, nullptr, act, Mrows, MLPd, Edim, 2);

  // 9) fc2 GEMM + residual -> d_out fp32
  gemm_bf16_kernel<<<(Mrows / 64) * (Edim / 32) / 8, 256, 0, stream>>>(
      act, w2, fc2_b, x1, out, nullptr, Mrows, Edim, MLPd, 1);
}
